// Nonlocal_pseudo_20255065768220
// MI455X (gfx1250) — matrix operand order repaired and checked by reading, not compiled, not run on hardware
//
#include <hip/hip_runtime.h>

// ---------------------------------------------------------------------------
// Non-local pseudo block, MI455X (gfx1250, wave32, WMMA).
// f = lam/(2*pi*dist^3) [B,N,N] materialized bf16 (128MB in d_ws),
// GEMMs via v_wmma_f32_16x16x32_bf16 (f32 accumulate).
// HBM traffic ~ 384MB => ~16us @ 23.3 TB/s; compute ~22 GFLOP bf16 WMMA.
// iter_kernel uses a 4-way K-split + LDS reduction so the f-streaming GEMM
// runs at ~4 waves/SIMD instead of 1 (latency hiding while streaming f).
// ---------------------------------------------------------------------------

typedef __bf16 bf16;
typedef __bf16 v16bf __attribute__((ext_vector_type(16)));
typedef float  v8f   __attribute__((ext_vector_type(8)));

#define LAM_FACTOR 0.015915494309189535f   // 0.1 * 1/(2*pi)  (typ==2, S=0.5, N_DIM=2)
#define INV_N      (1.0f/4096.0f)
#define HSTEP      0.05f
#define BN_EPS     1e-3f

union FragBF { v16bf v; uint4 q[2]; };

__device__ __forceinline__ v8f wmma_bf16(v16bf a, v16bf b, v8f c) {
  // (neg_a, A, neg_b, B, c_mod, C, reuse_a, reuse_b)
  return __builtin_amdgcn_wmma_f32_16x16x32_bf16(false, a, false, b, (short)0, c,
                                                 false, false);
}

// A-fragment (16x32 bf16, M x K), operand row-major, leading dim ld (halves).
// ISA 7.12.2: lo lanes K {k0+0..7, k0+16..23}, hi lanes {k0+8..15, k0+24..31}
// -> two contiguous 16B loads per lane.
__device__ __forceinline__ v16bf load_fragA(const bf16* base, int ld, int row0,
                                            int k0, int lane) {
  int r = lane & 15, hi = (lane >> 4) & 1;
  const bf16* p = base + (size_t)(row0 + r) * ld + k0 + hi * 8;
  FragBF fr;
  fr.q[0] = *(const uint4*)p;
  fr.q[1] = *(const uint4*)(p + 16);
  return fr.v;
}

// B-fragment (32x16 bf16, K x N) from the TRANSPOSED operand Bt[n][k]
// (row-major, ld in halves). Lane = column n0+r; lo lanes K {k0+0..7, k0+16..23},
// hi lanes {k0+8..15, k0+24..31} -> two contiguous 16B loads per lane.
__device__ __forceinline__ v16bf load_fragB_T(const bf16* baseT, int ld, int n0,
                                              int k0, int lane) {
  int r = lane & 15, hi = (lane >> 4) & 1;
  const bf16* p = baseT + (size_t)(n0 + r) * ld + k0 + hi * 8;
  FragBF fr;
  fr.q[0] = *(const uint4*)p;
  fr.q[1] = *(const uint4*)(p + 16);
  return fr.v;
}

// ---------------------------------------------------------------------------
// K0: WT[i][d][c] = W_ws[i][c][d] bf16; BN scale = gamma/sqrt(var+eps).
// ---------------------------------------------------------------------------
__global__ void prep_params(const float* __restrict__ W_ws,
                            const float* __restrict__ gammas,
                            const float* __restrict__ variances,
                            bf16* __restrict__ WT, float* __restrict__ scale) {
  int t = threadIdx.x;
  for (int idx = t; idx < 2 * 64 * 64; idx += 256) {
    int i = idx >> 12, rem = idx & 4095, d = rem >> 6, c = rem & 63;
    WT[idx] = (bf16)W_ws[i * 4096 + c * 64 + d];
  }
  for (int idx = t; idx < 128; idx += 256) {
    scale[idx] = gammas[idx] * rsqrtf(variances[idx] + BN_EPS);
  }
}

// ---------------------------------------------------------------------------
// K1: theta/phi projections (f32 -> bf16), bf16-consistent row norms,
// x transposed to gT0[b][c][n] bf16.
// ---------------------------------------------------------------------------
__global__ void proj_kernel(const float* __restrict__ x,
                            const float* __restrict__ tw_g,
                            const float* __restrict__ tb,
                            const float* __restrict__ pw_g,
                            const float* __restrict__ pb,
                            bf16* __restrict__ theta, bf16* __restrict__ phi,
                            float* __restrict__ rnT, float* __restrict__ rnP,
                            bf16* __restrict__ gT0) {
  __shared__ float tw[2048], pw[2048];
  int t = threadIdx.x;
  for (int i = t; i < 2048; i += 256) { tw[i] = tw_g[i]; pw[i] = pw_g[i]; }
  __syncthreads();

  int n = blockIdx.x * 256 + t;          // global pixel 0..16383
  int b = n >> 12, nl = n & 4095;
  float xr[64];
  const float* xp = x + (size_t)n * 64;
#pragma unroll
  for (int c = 0; c < 64; ++c) xr[c] = xp[c];

  float rt = 0.f, rp = 0.f;
  for (int d = 0; d < 32; ++d) {
    float at = tb[d], ap = pb[d];
#pragma unroll
    for (int c = 0; c < 64; ++c) {
      at += xr[c] * tw[c * 32 + d];
      ap += xr[c] * pw[c * 32 + d];
    }
    bf16 bt = (bf16)at, bp = (bf16)ap;
    theta[(size_t)n * 32 + d] = bt;
    phi[(size_t)n * 32 + d] = bp;
    float ft = (float)bt, fp = (float)bp;
    rt += ft * ft;
    rp += fp * fp;
  }
  rnT[n] = rt;
  rnP[n] = rp;

  bf16* g = gT0 + (size_t)b * 64 * 4096 + nl;
#pragma unroll
  for (int c = 0; c < 64; ++c) g[(size_t)c * 4096] = (bf16)xr[c];
}

// ---------------------------------------------------------------------------
// K2: f[b][m][n] = LAM_FACTOR / dist^3, one 16x16 tile per wave via a single
// K=32 bf16 WMMA (theta @ phi^T).
// ---------------------------------------------------------------------------
__global__ void dist_kernel(const bf16* __restrict__ theta,
                            const bf16* __restrict__ phi,
                            const float* __restrict__ rnT,
                            const float* __restrict__ rnP,
                            bf16* __restrict__ f) {
  int lane = threadIdx.x & 31, wave = threadIdx.x >> 5;
  int t = blockIdx.x * 8 + wave;         // tile id 0..262143
  int b = t >> 16;
  int rem = t & 65535;
  int m0 = (rem >> 8) << 4;
  int n0 = (rem & 255) << 4;
  int r = lane & 15, hi = (lane >> 4) & 1;

  v16bf A = load_fragA(theta + (size_t)b * 4096 * 32, 32, m0, 0, lane);
  v16bf B = load_fragB_T(phi + (size_t)b * 4096 * 32, 32, n0, 0, lane);
  v8f c = {};
  c = wmma_bf16(A, B, c);

  int col = n0 + r;
  float rb = rnP[b * 4096 + col];
  bf16* fb = f + (size_t)b * 4096 * 4096;
#pragma unroll
  for (int v = 0; v < 8; ++v) {
    int row = m0 + v + hi * 8;
    float d2 = rnT[b * 4096 + row] - 2.0f * c[v] + rb;
    d2 = fmaxf(d2, 1e-12f);
    float d = sqrtf(d2);
    float fp3 = d * d * d;               // never 0 after clamp -> divide_no_nan trivial
    fb[(size_t)row * 4096 + col] = (bf16)(LAM_FACTOR / fp3);
  }
}

// ---------------------------------------------------------------------------
// K3: f_sum over columns from the bf16-rounded f (consistent with f@g).
// One wave per row, b128 loads + shfl-xor reduce.
// ---------------------------------------------------------------------------
__global__ void fsum_kernel(const bf16* __restrict__ f, float* __restrict__ fsum) {
  int lane = threadIdx.x & 31, wave = threadIdx.x >> 5;
  int row = blockIdx.x * 8 + wave;       // global row (includes batch)
  const bf16* p = f + (size_t)row * 4096;
  float s = 0.f;
  for (int j = 0; j < 16; ++j) {
    uint4 q = *(const uint4*)(p + j * 256 + lane * 8);
    const bf16* h = (const bf16*)&q;
#pragma unroll
    for (int e = 0; e < 8; ++e) s += (float)h[e];
  }
#pragma unroll
  for (int m = 16; m >= 1; m >>= 1) s += __shfl_xor(s, m, 32);
  if (lane == 0) fsum[row] = s;
}

// ---------------------------------------------------------------------------
// K4: one iteration. Block = 8 waves = 2 row-tiles x 4 K-splits:
//   phase 1: each wave: partial fg = f[:, kseg] @ g[kseg, :]   (32 x 4 WMMA)
//   phase 2: LDS f32 reduction across the 4 K-splits (lane-wise, no relayout)
//   phase 3: per-wave fragment: fg' = (g*f_sum - fg)/N -> bf16 LDS tile
//   phase 4: y = fg' @ W^T (+bias, BN, relu), out = x + 0.05*y
//            -> d_out (f32) and transposed g_out (bf16)
// ---------------------------------------------------------------------------
__global__ void iter_kernel(const bf16* __restrict__ f,
                            const float* __restrict__ fsum,
                            const bf16* __restrict__ g_in,
                            bf16* __restrict__ g_out,
                            const float* __restrict__ x,
                            const bf16* __restrict__ WT,
                            const float* __restrict__ Wb,
                            const float* __restrict__ mean,
                            const float* __restrict__ beta,
                            const float* __restrict__ scale,
                            float* __restrict__ out) {
  __shared__ float accs[8][4][8][32];    // 32 KB: per-wave partial C fragments
  __shared__ bf16 tile2[2][16 * 80];     // 5 KB: fg' staging (80-half stride, 16B aligned)

  int lane = threadIdx.x & 31, wave = threadIdx.x >> 5;
  int r = lane & 15, hi = (lane >> 4) & 1;
  int rt = wave >> 2;                    // row tile within block (0/1)
  int ks = wave & 3;                     // K split (0..3)
  int rb_g = blockIdx.x * 32 + rt * 16;  // global row base of this row tile
  int b = rb_g >> 12;
  int rl = rb_g & 4095;
  const bf16* gb = g_in + (size_t)b * 64 * 4096;
  const bf16* frow = f + (size_t)rb_g * 4096;

  // phase 1: partial fg over K segment [ks*1024, ks*1024+1024)
  v8f acc[4] = {{}, {}, {}, {}};
  int kbeg = ks << 10, kend = kbeg + 1024;
  for (int k0 = kbeg; k0 < kend; k0 += 32) {
    // stream-prefetch f ~2KB ahead (global_prefetch_b8; dropped past the end)
    __builtin_prefetch(frow + (size_t)r * 4096 + k0 + 1024, 0, 1);
    v16bf A = load_fragA(frow, 4096, 0, k0, lane);
#pragma unroll
    for (int cf = 0; cf < 4; ++cf) {
      v16bf B = load_fragB_T(gb, 4096, cf * 16, k0, lane);
      acc[cf] = wmma_bf16(A, B, acc[cf]);
    }
  }
#pragma unroll
  for (int cf = 0; cf < 4; ++cf)
#pragma unroll
    for (int v = 0; v < 8; ++v) accs[wave][cf][v][lane] = acc[cf][v];
  __syncthreads();

  // phase 2: combine the 4 K-splits; this wave now owns channel-fragment df
  int df = ks;
  int ch = df * 16 + r;
  float comb[8];
#pragma unroll
  for (int v = 0; v < 8; ++v) {
    comb[v] = accs[rt * 4 + 0][df][v][lane] + accs[rt * 4 + 1][df][v][lane] +
              accs[rt * 4 + 2][df][v][lane] + accs[rt * 4 + 3][df][v][lane];
  }

  // phase 3: Laplacian normalization -> bf16 staging tile
  const bf16* gcol = gb + (size_t)ch * 4096 + rl;
#pragma unroll
  for (int v = 0; v < 8; ++v) {
    int rr = v + hi * 8;
    float gv = (float)gcol[rr];
    float fsv = fsum[rb_g + rr];
    float fg = (gv * fsv - comb[v]) * INV_N;
    tile2[rt][rr * 80 + ch] = (bf16)fg;
  }
  __syncthreads();

  // phase 4: y = fg' @ W^T (K=64 -> 2 WMMAs per output fragment)
  v16bf A0 = load_fragA(&tile2[rt][0], 80, 0, 0, lane);
  v16bf A1 = load_fragA(&tile2[rt][0], 80, 0, 32, lane);
  v16bf Bw0 = load_fragB_T(WT, 64, df * 16, 0, lane);
  v16bf Bw1 = load_fragB_T(WT, 64, df * 16, 32, lane);
  v8f yy = {};
  yy = wmma_bf16(A0, Bw0, yy);
  yy = wmma_bf16(A1, Bw1, yy);

  float wb = Wb[ch], mn = mean[ch], bt = beta[ch], sc = scale[ch];
#pragma unroll
  for (int v = 0; v < 8; ++v) {
    int row_g = rb_g + v + hi * 8;
    float yv = yy[v] + wb;
    yv = (yv - mn) * sc + bt;            // inference BN
    yv = fmaxf(yv, 0.f);                 // relu
    float o = x[(size_t)row_g * 64 + ch] + HSTEP * yv;
    out[(size_t)row_g * 64 + ch] = o;
    g_out[(size_t)b * 64 * 4096 + (size_t)ch * 4096 + rl + v + hi * 8] = (bf16)o;
  }
}

// ---------------------------------------------------------------------------
extern "C" void kernel_launch(void* const* d_in, const int* in_sizes, int n_in,
                              void* d_out, int out_size, void* d_ws,
                              size_t ws_size, hipStream_t stream) {
  (void)in_sizes; (void)n_in; (void)out_size; (void)ws_size;
  const float* x       = (const float*)d_in[0];
  const float* theta_w = (const float*)d_in[1];
  const float* theta_b = (const float*)d_in[2];
  const float* phi_w   = (const float*)d_in[3];
  const float* phi_b   = (const float*)d_in[4];
  const float* W_ws    = (const float*)d_in[5];
  const float* W_bs    = (const float*)d_in[6];
  const float* gammas  = (const float*)d_in[7];
  const float* betas   = (const float*)d_in[8];
  const float* means   = (const float*)d_in[9];
  const float* vars    = (const float*)d_in[10];
  float* out = (float*)d_out;

  char* ws = (char*)d_ws;
  bf16*  theta = (bf16*)(ws + 0);                               // 1 MB
  bf16*  phi   = (bf16*)(ws + (1u << 20));                      // 1 MB
  float* rnT   = (float*)(ws + (2u << 20));                     // 64 KB
  float* rnP   = (float*)(ws + (2u << 20) + (64u << 10));       // 64 KB
  float* fsum  = (float*)(ws + (2u << 20) + (128u << 10));      // 64 KB
  bf16*  WT    = (bf16*)(ws + (2u << 20) + (192u << 10));       // 16 KB
  float* scale = (float*)(ws + (2u << 20) + (208u << 10));      // 512 B
  bf16*  gA    = (bf16*)(ws + (3u << 20));                      // 4 MB
  bf16*  gB    = (bf16*)(ws + (7u << 20));                      // 4 MB
  bf16*  fbuf  = (bf16*)(ws + (16u << 20));                     // 128 MB

  prep_params<<<1, 256, 0, stream>>>(W_ws, gammas, vars, WT, scale);
  proj_kernel<<<64, 256, 0, stream>>>(x, theta_w, theta_b, phi_w, phi_b, theta,
                                      phi, rnT, rnP, gA);
  dist_kernel<<<32768, 256, 0, stream>>>(theta, phi, rnT, rnP, fbuf);
  fsum_kernel<<<2048, 256, 0, stream>>>(fbuf, fsum);
  iter_kernel<<<512, 256, 0, stream>>>(fbuf, fsum, gA, gB, x, WT, W_bs, means,
                                       betas, scale, out);
  iter_kernel<<<512, 256, 0, stream>>>(fbuf, fsum, gB, gA, x, WT + 4096,
                                       W_bs + 64, means + 64, betas + 64,
                                       scale + 64, out);
}
